// LinearMixtralSparseMoeBlock_45758581571880
// MI455X (gfx1250) — matrix operand order repaired and checked by reading, not compiled, not run on hardware
//
#include <hip/hip_runtime.h>
#include <hip/hip_bf16.h>

// ---------------------------------------------------------------------------
// Mixtral sparse MoE block for gfx1250 (MI455X), bf16 WMMA path, v3.
// Shapes fixed by the reference: B=2,S=2048 -> T=4096, D=1024, F=3584, E=8, K=2
// v3: double-buffered LDS + register prefetch so global weight streaming
//     (352 MB fp32, the roofline term) overlaps the WMMA chains.
// ---------------------------------------------------------------------------

typedef __attribute__((ext_vector_type(16))) __bf16 v16bf;
typedef __attribute__((ext_vector_type(8)))  __bf16 v8bf;
typedef __attribute__((ext_vector_type(4)))  __bf16 v4bf;
typedef __attribute__((ext_vector_type(8)))  float  v8f;
typedef __attribute__((ext_vector_type(4)))  float  v4f;

#define MOE_T 4096
#define MOE_D 1024
#define MOE_F 3584
#define MOE_E 8
#define KT    32          // WMMA K step (bf16 16x16x32)
#define MT    32          // M rows per workgroup (2 x 16 sub-tiles)
#define BN    128         // N columns per workgroup (8 waves x 16)
#define BKS   (KT + 8)    // transposed-B LDS row stride (40 elems = 80B, 16B-aligned)
#define HROWS 8448        // sum_e roundup(count_e,32) <= 2T + E*31 = 8440 -> pad

// ---------------------------------------------------------------------------
// 1) Router: one wave (32 lanes) per token. fp32 logits, softmax, top-2,
//    renormalize, append (token, slot, weight) to per-expert lists.
// ---------------------------------------------------------------------------
__global__ __launch_bounds__(256) void moe_router(
    const float* __restrict__ x, const float* __restrict__ rw,
    int* __restrict__ counts, int* __restrict__ tok,
    int* __restrict__ slot, float* __restrict__ wt)
{
    int gthread = blockIdx.x * blockDim.x + threadIdx.x;
    int t    = gthread >> 5;          // token id (one wave32 per token)
    int lane = threadIdx.x & 31;
    if (t >= MOE_T) return;

    const float* xr = x + (size_t)t * MOE_D;
    float acc[MOE_E];
#pragma unroll
    for (int e = 0; e < MOE_E; ++e) acc[e] = 0.0f;

    for (int d = lane; d < MOE_D; d += 32) {
        float xv = xr[d];
#pragma unroll
        for (int e = 0; e < MOE_E; ++e)
            acc[e] += xv * rw[e * MOE_D + d];
    }
#pragma unroll
    for (int e = 0; e < MOE_E; ++e) {
#pragma unroll
        for (int off = 16; off > 0; off >>= 1)
            acc[e] += __shfl_xor(acc[e], off, 32);
    }

    if (lane == 0) {
        float m = acc[0];
#pragma unroll
        for (int e = 1; e < MOE_E; ++e) m = fmaxf(m, acc[e]);
        float p[MOE_E]; float s = 0.0f;
#pragma unroll
        for (int e = 0; e < MOE_E; ++e) { p[e] = __expf(acc[e] - m); s += p[e]; }
        float inv = __builtin_amdgcn_rcpf(s);
#pragma unroll
        for (int e = 0; e < MOE_E; ++e) p[e] *= inv;
        // top-2 (first index wins ties, matching lax.top_k)
        int i0 = 0;
#pragma unroll
        for (int e = 1; e < MOE_E; ++e) if (p[e] > p[i0]) i0 = e;
        int i1 = (i0 == 0) ? 1 : 0;
#pragma unroll
        for (int e = 0; e < MOE_E; ++e)
            if (e != i0 && p[e] > p[i1]) i1 = e;
        float w0 = p[i0], w1 = p[i1];
        float rs = 1.0f / (w0 + w1);   // accurate: feeds final output scale
        w0 *= rs; w1 *= rs;

        int p0 = atomicAdd(&counts[i0], 1);
        tok[i0 * MOE_T + p0] = t; slot[i0 * MOE_T + p0] = 0; wt[i0 * MOE_T + p0] = w0;
        int p1 = atomicAdd(&counts[i1], 1);
        tok[i1 * MOE_T + p1] = t; slot[i1 * MOE_T + p1] = 1; wt[i1 * MOE_T + p1] = w1;
    }
}

// ---------------------------------------------------------------------------
// 2) Tiny exclusive scan of per-expert counts, padded to MT-row tiles.
// ---------------------------------------------------------------------------
__global__ void moe_scan(const int* __restrict__ counts, int* __restrict__ offsets)
{
    if (threadIdx.x == 0 && blockIdx.x == 0) {
        int o = 0;
        for (int e = 0; e < MOE_E; ++e) {
            offsets[e] = o;
            o += (counts[e] + (MT - 1)) & ~(MT - 1);
        }
    }
}

// ---------------------------------------------------------------------------
// WMMA fragment helpers (layouts from CDNA5 ISA 7.12.2, wave32).
// A (16-bit, 16x32): lane<16 -> M=lane, K = {0..7,16..23}; lane>=16 -> +8.
//   Row-major [.,KT] LDS tile: both K octets are 16B-contiguous -> 2x b128.
// B (16-bit, 32x16): lane<16 -> N=lane, K = {0..7,16..23}; lane>=16 -> +8.
//   Staged transposed [N][K] (stride BKS): both K octets 16B-contiguous -> 2x b128.
// C/D (f32 16x16):   lane<16 -> N=lane, VGPR i = M=i; lane>=16 -> M=i+8.
// ---------------------------------------------------------------------------
__device__ __forceinline__ v16bf fragA(const __bf16* sA, int lane, int rowbase)
{
    int row  = rowbase + (lane & 15);
    int koff = (lane >> 4) * 8;
    const __bf16* p = sA + row * KT + koff;
    v8bf lo = *(const v8bf*)(p);        // K koff .. koff+7
    v8bf hi = *(const v8bf*)(p + 16);   // K koff+16 .. koff+23
    return __builtin_shufflevector(lo, hi, 0,1,2,3,4,5,6,7,8,9,10,11,12,13,14,15);
}

__device__ __forceinline__ v16bf fragB(const __bf16* sB, int wave, int lane)
{
    int col  = wave * 16 + (lane & 15);
    int koff = (lane >> 4) * 8;
    const __bf16* p = sB + col * BKS + koff;
    v8bf lo = *(const v8bf*)(p);
    v8bf hi = *(const v8bf*)(p + 16);
    return __builtin_shufflevector(lo, hi, 0,1,2,3,4,5,6,7,8,9,10,11,12,13,14,15);
}

__device__ __forceinline__ v4bf cvt4(float a, float b, float c, float d)
{
    v4bf r = { (__bf16)a, (__bf16)b, (__bf16)c, (__bf16)d };
    return r;
}

// ---------------------------------------------------------------------------
// 3) Fused gate+up GEMM: h = silu(x @ Wg) * (x @ Wu), bf16 out.
//    grid = (F/BN, T/MT, E); 256 threads = 8 waves; wave w -> cols w*16..+16.
//    Double-buffered LDS; per-iteration: prefetch(k+1) -> WMMA(k) -> store(k+1).
// ---------------------------------------------------------------------------
__global__ __launch_bounds__(256) void moe_gateup(
    const float* __restrict__ x,
    const float* __restrict__ wg, const float* __restrict__ wu,
    const int* __restrict__ counts, const int* __restrict__ offsets,
    const int* __restrict__ tok, __bf16* __restrict__ hbuf)
{
    int e  = blockIdx.z;
    int mt = blockIdx.y;
    int nb = blockIdx.x * BN;
    int cnt = counts[e];
    if (mt * MT >= cnt) return;

    __shared__ __bf16 sA[2][MT * KT];      // 32x32 row-major
    __shared__ __bf16 sG[2][BN * BKS];     // transposed [N][K]
    __shared__ __bf16 sU[2][BN * BKS];
    __shared__ int    stok[MT];

    int tid  = threadIdx.x;
    int wave = tid >> 5;
    int lane = tid & 31;

    if (tid < MT) {
        int r = mt * MT + tid;
        stok[tid] = tok[e * MOE_T + ((r < cnt) ? r : (cnt - 1))];
    }
    __syncthreads();

    const float* wge = wg + (size_t)e * MOE_D * MOE_F;
    const float* wue = wu + (size_t)e * MOE_D * MOE_F;

    // per-thread staging coordinates
    const int aidx = tid * 4;                   // A: 4 consecutive elems
    const int ar   = aidx >> 5, ac = aidx & 31;
    const size_t arow = (size_t)stok[ar] * MOE_D + ac;

    v8f accG[2] = {{}, {}};
    v8f accU[2] = {{}, {}};

    // ---- prologue: stage tile k=0 into buffer 0
    {
        v4f v = *(const v4f*)(x + arow);
        *(v4bf*)(&sA[0][aidx]) = cvt4(v[0], v[1], v[2], v[3]);
#pragma unroll
        for (int it = 0; it < 4; ++it) {
            int s  = it * 256 + tid;
            int c  = s & 127;
            int rg = s >> 7;
            size_t base = (size_t)(rg * 4) * MOE_F + nb + c;
            *(v4bf*)(&sG[0][c * BKS + rg * 4]) =
                cvt4(wge[base], wge[base + MOE_F], wge[base + 2 * MOE_F], wge[base + 3 * MOE_F]);
            *(v4bf*)(&sU[0][c * BKS + rg * 4]) =
                cvt4(wue[base], wue[base + MOE_F], wue[base + 2 * MOE_F], wue[base + 3 * MOE_F]);
        }
    }
    __syncthreads();

    constexpr int NK = MOE_D / KT;             // 32 K-steps
    for (int kk = 0; kk < NK; ++kk) {
        const int cur = kk & 1, nxt = cur ^ 1;
        const int k1  = (kk + 1) * KT;
        const bool has_next = (kk + 1) < NK;

        // ---- issue next tile's global loads first (overlap with WMMA)
        v4f pa;
        float pg[16], pu[16];
        if (has_next) {
            pa = *(const v4f*)(x + arow + k1);
#pragma unroll
            for (int it = 0; it < 4; ++it) {
                int s  = it * 256 + tid;
                int c  = s & 127;
                int rg = s >> 7;
                size_t base = (size_t)(k1 + rg * 4) * MOE_F + nb + c;
#pragma unroll
                for (int j = 0; j < 4; ++j) {
                    pg[it * 4 + j] = wge[base + (size_t)j * MOE_F];
                    pu[it * 4 + j] = wue[base + (size_t)j * MOE_F];
                }
            }
        }

        // ---- WMMA on current buffer
        v16bf a0 = fragA(sA[cur], lane, 0);
        v16bf a1 = fragA(sA[cur], lane, 16);
        v16bf bg = fragB(sG[cur], wave, lane);
        v16bf bu = fragB(sU[cur], wave, lane);
        accG[0] = __builtin_amdgcn_wmma_f32_16x16x32_bf16(false, a0, false, bg,
                                                          (short)0, accG[0], false, false);
        accG[1] = __builtin_amdgcn_wmma_f32_16x16x32_bf16(false, a1, false, bg,
                                                          (short)0, accG[1], false, false);
        accU[0] = __builtin_amdgcn_wmma_f32_16x16x32_bf16(false, a0, false, bu,
                                                          (short)0, accU[0], false, false);
        accU[1] = __builtin_amdgcn_wmma_f32_16x16x32_bf16(false, a1, false, bu,
                                                          (short)0, accU[1], false, false);

        // ---- convert + store prefetched tile into the other buffer
        if (has_next) {
            *(v4bf*)(&sA[nxt][aidx]) = cvt4(pa[0], pa[1], pa[2], pa[3]);
#pragma unroll
            for (int it = 0; it < 4; ++it) {
                int s  = it * 256 + tid;
                int c  = s & 127;
                int rg = s >> 7;
                *(v4bf*)(&sG[nxt][c * BKS + rg * 4]) =
                    cvt4(pg[it * 4], pg[it * 4 + 1], pg[it * 4 + 2], pg[it * 4 + 3]);
                *(v4bf*)(&sU[nxt][c * BKS + rg * 4]) =
                    cvt4(pu[it * 4], pu[it * 4 + 1], pu[it * 4 + 2], pu[it * 4 + 3]);
            }
        }
        __syncthreads();
    }

    // ---- epilogue: h = silu(g) * u -> bf16, compact rows at offsets[e]
    int off   = offsets[e];
    int col   = nb + wave * 16 + (lane & 15);
    int mbase = (lane >> 4) << 3;
    bool full = (mt * MT + MT) <= cnt;
    if (full) {
#pragma unroll
        for (int sub = 0; sub < 2; ++sub) {
#pragma unroll
            for (int i = 0; i < 8; ++i) {
                int r = mt * MT + sub * 16 + mbase + i;
                float g = accG[sub][i], u = accU[sub][i];
                float h = g * __builtin_amdgcn_rcpf(1.0f + __expf(-g)) * u;
                hbuf[(size_t)(off + r) * MOE_F + col] = (__bf16)h;
            }
        }
    } else {
#pragma unroll
        for (int sub = 0; sub < 2; ++sub) {
#pragma unroll
            for (int i = 0; i < 8; ++i) {
                int r = mt * MT + sub * 16 + mbase + i;
                if (r < cnt) {
                    float g = accG[sub][i], u = accU[sub][i];
                    float h = g * __builtin_amdgcn_rcpf(1.0f + __expf(-g)) * u;
                    hbuf[(size_t)(off + r) * MOE_F + col] = (__bf16)h;
                }
            }
        }
    }
}

// ---------------------------------------------------------------------------
// 4) Down projection: partial[token][slot] = w * (h @ Wd). No atomics:
//    each token has exactly one slot-0 and one slot-1 contribution.
// ---------------------------------------------------------------------------
__global__ __launch_bounds__(256) void moe_down(
    const __bf16* __restrict__ hbuf, const float* __restrict__ wd,
    const int* __restrict__ counts, const int* __restrict__ offsets,
    const int* __restrict__ tok, const int* __restrict__ slot,
    const float* __restrict__ wt, float* __restrict__ partial)
{
    int e  = blockIdx.z;
    int mt = blockIdx.y;
    int nb = blockIdx.x * BN;
    int cnt = counts[e];
    if (mt * MT >= cnt) return;

    __shared__ __bf16 sA[2][MT * KT];
    __shared__ __bf16 sB[2][BN * BKS];
    __shared__ int    stok[MT];
    __shared__ int    sslot[MT];
    __shared__ float  swt[MT];

    int tid  = threadIdx.x;
    int wave = tid >> 5;
    int lane = tid & 31;

    if (tid < MT) {
        int r  = mt * MT + tid;
        int rc = (r < cnt) ? r : (cnt - 1);
        stok[tid]  = tok[e * MOE_T + rc];
        sslot[tid] = slot[e * MOE_T + rc];
        swt[tid]   = wt[e * MOE_T + rc];
    }

    const float* wde = wd + (size_t)e * MOE_F * MOE_D;
    int rowbase = offsets[e] + mt * MT;

    // A staging: 1024 bf16, 8/thread, first 128 threads
    const int aidx = tid * 8;
    const int ar = aidx >> 5, ac = aidx & 31;
    const size_t harow = (size_t)(rowbase + ar) * MOE_F + ac;

    v8f acc[2] = {{}, {}};

    // ---- prologue: stage tile k=0 into buffer 0
    if (tid < 128)
        *(v8bf*)(&sA[0][aidx]) = *(const v8bf*)(hbuf + harow);
#pragma unroll
    for (int it = 0; it < 4; ++it) {
        int s  = it * 256 + tid;
        int c  = s & 127;
        int rg = s >> 7;
        size_t base = (size_t)(rg * 4) * MOE_D + nb + c;
        *(v4bf*)(&sB[0][c * BKS + rg * 4]) =
            cvt4(wde[base], wde[base + MOE_D], wde[base + 2 * MOE_D], wde[base + 3 * MOE_D]);
    }
    __syncthreads();

    constexpr int NK = MOE_F / KT;             // 112 K-steps
    for (int kk = 0; kk < NK; ++kk) {
        const int cur = kk & 1, nxt = cur ^ 1;
        const int k1  = (kk + 1) * KT;
        const bool has_next = (kk + 1) < NK;

        v8bf pa = {};
        float pb[16];
        if (has_next) {
            if (tid < 128)
                pa = *(const v8bf*)(hbuf + harow + k1);
#pragma unroll
            for (int it = 0; it < 4; ++it) {
                int s  = it * 256 + tid;
                int c  = s & 127;
                int rg = s >> 7;
                size_t base = (size_t)(k1 + rg * 4) * MOE_D + nb + c;
#pragma unroll
                for (int j = 0; j < 4; ++j)
                    pb[it * 4 + j] = wde[base + (size_t)j * MOE_D];
            }
        }

        v16bf a0 = fragA(sA[cur], lane, 0);
        v16bf a1 = fragA(sA[cur], lane, 16);
        v16bf b  = fragB(sB[cur], wave, lane);
        acc[0] = __builtin_amdgcn_wmma_f32_16x16x32_bf16(false, a0, false, b,
                                                         (short)0, acc[0], false, false);
        acc[1] = __builtin_amdgcn_wmma_f32_16x16x32_bf16(false, a1, false, b,
                                                         (short)0, acc[1], false, false);

        if (has_next) {
            if (tid < 128)
                *(v8bf*)(&sA[nxt][aidx]) = pa;
#pragma unroll
            for (int it = 0; it < 4; ++it) {
                int s  = it * 256 + tid;
                int c  = s & 127;
                int rg = s >> 7;
                *(v4bf*)(&sB[nxt][c * BKS + rg * 4]) =
                    cvt4(pb[it * 4], pb[it * 4 + 1], pb[it * 4 + 2], pb[it * 4 + 3]);
            }
        }
        __syncthreads();
    }

    int col   = nb + wave * 16 + (lane & 15);
    int mbase = (lane >> 4) << 3;
    bool full = (mt * MT + MT) <= cnt;
    if (full) {
#pragma unroll
        for (int sub = 0; sub < 2; ++sub) {
#pragma unroll
            for (int i = 0; i < 8; ++i) {
                int mrow = sub * 16 + mbase + i;
                size_t t = (size_t)stok[mrow];
                partial[(t * 2 + sslot[mrow]) * MOE_D + col] = acc[sub][i] * swt[mrow];
            }
        }
    } else {
#pragma unroll
        for (int sub = 0; sub < 2; ++sub) {
#pragma unroll
            for (int i = 0; i < 8; ++i) {
                int mrow = sub * 16 + mbase + i;
                int r = mt * MT + mrow;
                if (r < cnt) {
                    size_t t = (size_t)stok[mrow];
                    partial[(t * 2 + sslot[mrow]) * MOE_D + col] = acc[sub][i] * swt[mrow];
                }
            }
        }
    }
}

// ---------------------------------------------------------------------------
// 5) Final combine: out = partial[:,0,:] + partial[:,1,:]  (vectorized x4)
// ---------------------------------------------------------------------------
__global__ __launch_bounds__(256) void moe_combine(
    const float* __restrict__ partial, float* __restrict__ out)
{
    size_t i4 = (size_t)blockIdx.x * blockDim.x + threadIdx.x;  // over T*D/4
    if (i4 >= (size_t)MOE_T * MOE_D / 4) return;
    size_t i = i4 * 4;
    size_t t = i / MOE_D, d = i - t * MOE_D;
    v4f a = *(const v4f*)(partial + (t * 2 + 0) * MOE_D + d);
    v4f b = *(const v4f*)(partial + (t * 2 + 1) * MOE_D + d);
    *(v4f*)(out + i) = a + b;
}

// ---------------------------------------------------------------------------
extern "C" void kernel_launch(void* const* d_in, const int* in_sizes, int n_in,
                              void* d_out, int out_size, void* d_ws, size_t ws_size,
                              hipStream_t stream)
{
    const float* x  = (const float*)d_in[0];   // [2,2048,1024]
    const float* rw = (const float*)d_in[1];   // [8,1024]
    const float* wg = (const float*)d_in[2];   // [8,1024,3584]
    const float* wu = (const float*)d_in[3];   // [8,1024,3584]
    const float* wd = (const float*)d_in[4];   // [8,3584,1024]
    float* out = (float*)d_out;                // [2,2048,1024]
    char*  ws  = (char*)d_ws;

    // workspace layout (256B aligned sections)
    size_t o = 0;
    int*   counts  = (int*)(ws + o);  o += 256;                       // E ints
    int*   offsets = (int*)(ws + o);  o += 256;                       // E ints
    int*   tok     = (int*)(ws + o);  o += (size_t)MOE_E * MOE_T * 4; // 128 KB
    int*   slot    = (int*)(ws + o);  o += (size_t)MOE_E * MOE_T * 4;
    float* wt      = (float*)(ws + o);o += (size_t)MOE_E * MOE_T * 4;
    o = (o + 255) & ~(size_t)255;
    __bf16* hbuf   = (__bf16*)(ws + o); o += (size_t)HROWS * MOE_F * 2; // ~60.5 MB
    o = (o + 255) & ~(size_t)255;
    float* partial = (float*)(ws + o); o += (size_t)MOE_T * 2 * MOE_D * 4; // 32 MB
    (void)o; (void)ws_size; (void)in_sizes; (void)n_in; (void)out_size;

    hipMemsetAsync(counts, 0, MOE_E * sizeof(int), stream);

    // 1) router: one wave per token -> 4096 waves, 256 threads/block
    moe_router<<<(MOE_T * 32) / 256, 256, 0, stream>>>(x, rw, counts, tok, slot, wt);

    // 2) scan
    moe_scan<<<1, 32, 0, stream>>>(counts, offsets);

    // 3) gate+up fused GEMM
    dim3 g1(MOE_F / BN, MOE_T / MT, MOE_E);   // (28, 128, 8), early-exit per count
    moe_gateup<<<g1, 256, 0, stream>>>(x, wg, wu, counts, offsets, tok, hbuf);

    // 4) down projection
    dim3 g2(MOE_D / BN, MOE_T / MT, MOE_E);   // (8, 128, 8)
    moe_down<<<g2, 256, 0, stream>>>(hbuf, wd, counts, offsets, tok, slot, wt, partial);

    // 5) combine slots
    moe_combine<<<(int)(((size_t)MOE_T * MOE_D / 4 + 255) / 256), 256, 0, stream>>>(partial, out);
}
